// WindowAttention_9079560864359
// MI455X (gfx1250) — matrix operand order repaired and checked by reading, not compiled, not run on hardware
//
#include <hip/hip_runtime.h>

typedef _Float16 f16;
typedef __attribute__((ext_vector_type(16))) _Float16 v16h;
typedef __attribute__((ext_vector_type(4)))  _Float16 v4h;
typedef __attribute__((ext_vector_type(8)))  float    v8f;
typedef __attribute__((ext_vector_type(4)))  float    f32x4;
typedef __attribute__((ext_vector_type(4)))  unsigned int u32x4;
typedef int i32x4v __attribute__((vector_size(16)));

#define DIM   1024
#define NT    49
#define HEADS 16
#define HD    64
#define NWIN  2048
#define NMASK 64

#if __has_builtin(__builtin_amdgcn_global_load_async_to_lds_b128) && \
    __has_builtin(__builtin_amdgcn_s_wait_asynccnt)
#define USE_ASYNC_LDS 1
#else
#define USE_ASYNC_LDS 0
#endif

union Frag { v16h h; u32x4 q[2]; };

__device__ inline v8f vzero8() {
  v8f z;
  #pragma unroll
  for (int i = 0; i < 8; ++i) z[i] = 0.0f;
  return z;
}

// ---------------------------------------------------------------------------
// Kernel 0a: f32 -> f16 weight conversion (qkv_w, proj_w); weights live in L2.
// ---------------------------------------------------------------------------
__global__ void cvt_f32_f16(const float* __restrict__ src, f16* __restrict__ dst, int n) {
  int i = blockIdx.x * blockDim.x + threadIdx.x;
  if (i < n) dst[i] = (f16)src[i];
}

// ---------------------------------------------------------------------------
// Kernel 0b: combined bias table bm[wi][h][q][k] = rel_pos_bias + attn_mask
// (9.8 MB, L2-resident; removes div/mod-7 + scattered loads from the 32768
//  attention blocks)
// ---------------------------------------------------------------------------
#define BM_ELEMS (NMASK * HEADS * NT * NT)
__global__ void build_bias_mask(const float* __restrict__ bias_table,
                                const float* __restrict__ attn_mask,
                                float* __restrict__ bm) {
  int i = blockIdx.x * blockDim.x + threadIdx.x;
  if (i >= BM_ELEMS) return;
  int kk = i % NT;
  int q0 = (i / NT) % NT;
  int h  = (i / (NT * NT)) % HEADS;
  int wi = i / (NT * NT * HEADS);
  int qi = q0 / 7, qj = q0 % 7, ki = kk / 7, kj = kk % 7;
  int ridx = (qi - ki + 6) * 13 + (qj - kj + 6);
  bm[i] = bias_table[ridx * HEADS + h] + attn_mask[((size_t)wi * NT + q0) * NT + kk];
}

// ---------------------------------------------------------------------------
// Kernel 1: QKV GEMM per window; x window (64x1024 f16 = 128KB) staged once.
// Each of 8 waves: 4 M-tiles x 4 N-tiles accumulators, 6 N-chunks.
// Scatter: q scaled, k row-major [tok][d], v transposed [d][tok], all f16.
// ---------------------------------------------------------------------------
__global__ __launch_bounds__(256) void swin_qkv_gemm(
    const float* __restrict__ x, const f16* __restrict__ wh,
    const float* __restrict__ qkv_b,
    f16* __restrict__ qb, f16* __restrict__ kb, f16* __restrict__ vT)
{
  __shared__ f16 lx[64 * 1024];   // 128 KB of the WGP's 320 KB LDS
  const int b    = blockIdx.x;
  const int tid  = threadIdx.x;
  const int wave = tid >> 5;
  const int lr   = tid & 15;
  const int lhi  = (tid & 31) >> 4;

  // Zero padded token region of vT (tokens 49..63) so 0*pad stays 0 later.
  for (int i = tid; i < HEADS * HD * (64 - NT); i += 256) {
    int t = NT + (i % (64 - NT));
    int d = (i / (64 - NT)) % HD;
    int h = i / ((64 - NT) * HD);
    vT[(((size_t)b * HEADS + h) * HD + d) * 64 + t] = (f16)0.0f;
  }

  // Stage full x window f32 -> f16, zero-pad tokens >= 49.
  for (int i = tid; i < (64 * 1024) / 4; i += 256) {
    int e = i * 4, tok = e >> 10, kk = e & 1023;
    f32x4 f; f[0] = f[1] = f[2] = f[3] = 0.0f;
    if (tok < NT)
      f = *(const f32x4*)&x[((size_t)b * NT + tok) * DIM + kk];
    v4h hv; hv[0]=(f16)f[0]; hv[1]=(f16)f[1]; hv[2]=(f16)f[2]; hv[3]=(f16)f[3];
    *(v4h*)&lx[e] = hv;
  }
  __syncthreads();

  for (int nc = 0; nc < 6; ++nc) {
    v8f acc[16];
    for (int t = 0; t < 16; ++t) acc[t] = vzero8();

    for (int kt = 0; kt < 32; ++kt) {
      const int k0 = kt * 32;
      Frag a[4];
      #pragma unroll
      for (int mi = 0; mi < 4; ++mi) {
        const int row = mi * 16 + lr;
        a[mi].q[0] = *(const u32x4*)&lx[row * 1024 + k0 + lhi * 16];
        a[mi].q[1] = *(const u32x4*)&lx[row * 1024 + k0 + lhi * 16 + 8];
      }
      Frag bf[4];
      const int kg = k0 + lhi * 16;
      #pragma unroll
      for (int j = 0; j < 4; ++j) {
        const int n = (wave * 24 + nc * 4 + j) * 16 + lr;
        const u32x4* p = (const u32x4*)&wh[(size_t)n * DIM + kg];
        bf[j].q[0] = p[0]; bf[j].q[1] = p[1];
      }
      #pragma unroll
      for (int mi = 0; mi < 4; ++mi)
        #pragma unroll
        for (int j = 0; j < 4; ++j)
          acc[mi * 4 + j] = __builtin_amdgcn_wmma_f32_16x16x32_f16(
              false, a[mi].h, false, bf[j].h, (short)0, acc[mi * 4 + j], false, false);
    }

    // Epilogue: add bias, scatter into q / k / vT (f16).
    #pragma unroll
    for (int mi = 0; mi < 4; ++mi)
      #pragma unroll
      for (int j = 0; j < 4; ++j) {
        const int ncol = (wave * 24 + nc * 4 + j) * 16 + lr;
        const float bias = qkv_b[ncol];
        const int sec = ncol >> 10;
        const int h   = (ncol & 1023) >> 6;
        const int d   = ncol & 63;
        const size_t hb = (size_t)b * HEADS + h;
        #pragma unroll
        for (int r = 0; r < 8; ++r) {
          const int tok = mi * 16 + r + lhi * 8;
          if (tok >= NT) continue;
          const float val = acc[mi * 4 + j][r] + bias;
          if (sec == 0)      qb[(hb * 64 + tok) * HD + d] = (f16)(val * 0.125f); // q * hd^-0.5
          else if (sec == 1) kb[(hb * 64 + tok) * HD + d] = (f16)val;
          else               vT[(hb * HD + d) * 64 + tok] = (f16)val;
        }
      }
  }
}

// ---------------------------------------------------------------------------
// Kernel 2: attention per (window, head). Single wave per block.
// S = q k^T (32 WMMAs) + bias/mask folded at store -> LDS; LDS-only softmax;
// O = P v (32 WMMAs); write head-concatenated f16 output [b][tok pad64][1024].
// ---------------------------------------------------------------------------
__global__ __launch_bounds__(32) void swin_attn(
    const f16* __restrict__ qb, const f16* __restrict__ kb, const f16* __restrict__ vT,
    const float* __restrict__ bm, f16* __restrict__ oh)
{
  __shared__ float S[64 * 64];
  __shared__ f16   P[64 * 64];
  const int b   = blockIdx.x;
  const int h   = blockIdx.y;
  const int ln  = threadIdx.x;
  const int lr  = ln & 15;
  const int lhi = ln >> 4;
  const size_t base = ((size_t)b * HEADS + h) * 64 * HD;
  const f16* q = qb + base;
  const f16* k = kb + base;
  const f16* v = vT + base;
  const float* bmp = bm + ((size_t)(b & (NMASK - 1)) * HEADS + h) * NT * NT;

  // --- S = q @ k^T ---
  v8f acc[16];
  for (int t = 0; t < 16; ++t) acc[t] = vzero8();
  #pragma unroll
  for (int kt = 0; kt < 2; ++kt) {
    const int k0 = kt * 32;
    Frag a[4], bf[4];
    #pragma unroll
    for (int mi = 0; mi < 4; ++mi) {
      const int row = mi * 16 + lr;
      a[mi].q[0] = *(const u32x4*)&q[row * HD + k0 + lhi * 16];
      a[mi].q[1] = *(const u32x4*)&q[row * HD + k0 + lhi * 16 + 8];
    }
    #pragma unroll
    for (int ni = 0; ni < 4; ++ni) {
      const int n = ni * 16 + lr;
      const u32x4* p = (const u32x4*)&k[n * HD + k0 + lhi * 16];
      bf[ni].q[0] = p[0]; bf[ni].q[1] = p[1];
    }
    #pragma unroll
    for (int mi = 0; mi < 4; ++mi)
      #pragma unroll
      for (int ni = 0; ni < 4; ++ni)
        acc[mi * 4 + ni] = __builtin_amdgcn_wmma_f32_16x16x32_f16(
            false, a[mi].h, false, bf[ni].h, (short)0, acc[mi * 4 + ni], false, false);
  }
  // Store S with bias+mask folded in (parallel across all lanes).
  #pragma unroll
  for (int mi = 0; mi < 4; ++mi)
    #pragma unroll
    for (int ni = 0; ni < 4; ++ni)
      #pragma unroll
      for (int r = 0; r < 8; ++r) {
        const int row = mi * 16 + r + lhi * 8;
        const int col = ni * 16 + lr;
        float sv = acc[mi * 4 + ni][r];
        if (row < NT && col < NT) sv += bmp[row * NT + col];
        S[row * 64 + col] = sv;
      }
  __syncthreads();

  // --- LDS-only softmax over keys [0,49) ---
  for (int q0 = ln; q0 < 64; q0 += 32) {
    if (q0 >= NT) {
      for (int kk = 0; kk < 64; ++kk) P[q0 * 64 + kk] = (f16)0.0f;
      continue;
    }
    float mx = -1e30f;
    for (int kk = 0; kk < NT; ++kk) mx = fmaxf(mx, S[q0 * 64 + kk]);
    float sum = 0.0f;
    for (int kk = 0; kk < NT; ++kk) {
      float e = __expf(S[q0 * 64 + kk] - mx);
      sum += e;
      S[q0 * 64 + kk] = e;
    }
    const float inv = 1.0f / sum;
    for (int kk = 0; kk < 64; ++kk)
      P[q0 * 64 + kk] = (f16)((kk < NT) ? S[q0 * 64 + kk] * inv : 0.0f);
  }
  __syncthreads();

  // --- O = P @ v ---
  v8f oacc[16];
  for (int t = 0; t < 16; ++t) oacc[t] = vzero8();
  #pragma unroll
  for (int kt = 0; kt < 2; ++kt) {
    const int k0 = kt * 32;
    Frag a[4], bf[4];
    #pragma unroll
    for (int mi = 0; mi < 4; ++mi) {
      const int row = mi * 16 + lr;
      a[mi].q[0] = *(const u32x4*)&P[row * 64 + k0 + lhi * 16];
      a[mi].q[1] = *(const u32x4*)&P[row * 64 + k0 + lhi * 16 + 8];
    }
    #pragma unroll
    for (int ni = 0; ni < 4; ++ni) {
      const int n = ni * 16 + lr;                               // d-channel
      const u32x4* p = (const u32x4*)&v[n * 64 + k0 + lhi * 16]; // vT[d][tok]
      bf[ni].q[0] = p[0]; bf[ni].q[1] = p[1];
    }
    #pragma unroll
    for (int mi = 0; mi < 4; ++mi)
      #pragma unroll
      for (int ni = 0; ni < 4; ++ni)
        oacc[mi * 4 + ni] = __builtin_amdgcn_wmma_f32_16x16x32_f16(
            false, a[mi].h, false, bf[ni].h, (short)0, oacc[mi * 4 + ni], false, false);
  }
  #pragma unroll
  for (int mi = 0; mi < 4; ++mi)
    #pragma unroll
    for (int ni = 0; ni < 4; ++ni)
      #pragma unroll
      for (int r = 0; r < 8; ++r) {
        const int tok = mi * 16 + r + lhi * 8;
        if (tok < NT)
          oh[((size_t)b * 64 + tok) * DIM + h * HD + ni * 16 + lr] =
              (f16)oacc[mi * 4 + ni][r];
      }
}

// ---------------------------------------------------------------------------
// Kernel 3: projection GEMM per window: out = oh @ proj_w^T + proj_b (f32 out)
// oh window staged once (128KB) via async LDS copies when available.
// ---------------------------------------------------------------------------
__global__ __launch_bounds__(256) void swin_proj_gemm(
    const f16* __restrict__ oh, const f16* __restrict__ pwh,
    const float* __restrict__ proj_b, float* __restrict__ out)
{
  __shared__ f16 lo[64 * 1024];   // 128 KB
  const int b    = blockIdx.x;
  const int tid  = threadIdx.x;
  const int wave = tid >> 5;
  const int lr   = tid & 15;
  const int lhi  = (tid & 31) >> 4;

#if USE_ASYNC_LDS
  for (int i = tid; i < (64 * 1024) / 8; i += 256) {
    const int e = i * 8;
    __builtin_amdgcn_global_load_async_to_lds_b128(
        (__attribute__((address_space(1))) i32x4v*)&oh[(size_t)b * 64 * DIM + e],
        (__attribute__((address_space(3))) i32x4v*)&lo[e], 0, 0);
  }
  __builtin_amdgcn_s_wait_asynccnt(0);
  __syncthreads();
#else
  for (int i = tid; i < (64 * 1024) / 8; i += 256) {
    const int e = i * 8;
    *(u32x4*)&lo[e] = *(const u32x4*)&oh[(size_t)b * 64 * DIM + e];
  }
  __syncthreads();
#endif

  for (int nc = 0; nc < 2; ++nc) {
    v8f acc[16];
    for (int t = 0; t < 16; ++t) acc[t] = vzero8();

    for (int kt = 0; kt < 32; ++kt) {
      const int k0 = kt * 32;
      Frag a[4];
      #pragma unroll
      for (int mi = 0; mi < 4; ++mi) {
        const int row = mi * 16 + lr;
        a[mi].q[0] = *(const u32x4*)&lo[row * 1024 + k0 + lhi * 16];
        a[mi].q[1] = *(const u32x4*)&lo[row * 1024 + k0 + lhi * 16 + 8];
      }
      Frag bf[4];
      const int kg = k0 + lhi * 16;
      #pragma unroll
      for (int j = 0; j < 4; ++j) {
        const int n = (wave * 8 + nc * 4 + j) * 16 + lr;
        const u32x4* p = (const u32x4*)&pwh[(size_t)n * DIM + kg];
        bf[j].q[0] = p[0]; bf[j].q[1] = p[1];
      }
      #pragma unroll
      for (int mi = 0; mi < 4; ++mi)
        #pragma unroll
        for (int j = 0; j < 4; ++j)
          acc[mi * 4 + j] = __builtin_amdgcn_wmma_f32_16x16x32_f16(
              false, a[mi].h, false, bf[j].h, (short)0, acc[mi * 4 + j], false, false);
    }

    #pragma unroll
    for (int mi = 0; mi < 4; ++mi)
      #pragma unroll
      for (int j = 0; j < 4; ++j) {
        const int ncol = (wave * 8 + nc * 4 + j) * 16 + lr;
        const float bias = proj_b[ncol];
        #pragma unroll
        for (int r = 0; r < 8; ++r) {
          const int tok = mi * 16 + r + lhi * 8;
          if (tok < NT)
            out[((size_t)b * NT + tok) * DIM + ncol] = acc[mi * 4 + j][r] + bias;
        }
      }
  }
}

// ---------------------------------------------------------------------------
extern "C" void kernel_launch(void* const* d_in, const int* in_sizes, int n_in,
                              void* d_out, int out_size, void* d_ws, size_t ws_size,
                              hipStream_t stream) {
  const float* x          = (const float*)d_in[0];
  const float* attn_mask  = (const float*)d_in[1];
  const float* qkv_w      = (const float*)d_in[2];
  const float* qkv_b      = (const float*)d_in[3];
  const float* proj_w     = (const float*)d_in[4];
  const float* proj_b     = (const float*)d_in[5];
  const float* bias_table = (const float*)d_in[6];
  float* out = (float*)d_out;

  char* ws = (char*)d_ws;
  f16* wh  = (f16*)ws; ws += (size_t)3 * DIM * DIM * sizeof(f16);   // qkv_w f16 [3072][1024]
  f16* pwh = (f16*)ws; ws += (size_t)DIM * DIM * sizeof(f16);       // proj_w f16 [1024][1024]
  const size_t hsz = (size_t)NWIN * HEADS * 64 * HD * sizeof(f16);  // 256 MB each
  f16* qb = (f16*)ws; ws += hsz;                                    // [b][h][tok pad64][d]
  f16* kb = (f16*)ws; ws += hsz;                                    // [b][h][tok pad64][d]
  f16* vT = (f16*)ws; ws += hsz;                                    // [b][h][d][tok pad64]
  f16* oh = (f16*)ws; ws += (size_t)NWIN * 64 * DIM * sizeof(f16);  // [b][tok pad64][dim]
  float* bm = (float*)ws; ws += (size_t)BM_ELEMS * sizeof(float);   // [wi][h][q][k]

  cvt_f32_f16<<<(3 * DIM * DIM + 255) / 256, 256, 0, stream>>>(qkv_w, wh, 3 * DIM * DIM);
  cvt_f32_f16<<<(DIM * DIM + 255) / 256, 256, 0, stream>>>(proj_w, pwh, DIM * DIM);
  build_bias_mask<<<(BM_ELEMS + 255) / 256, 256, 0, stream>>>(bias_table, attn_mask, bm);

  swin_qkv_gemm<<<NWIN, 256, 0, stream>>>(x, wh, qkv_b, qb, kb, vT);
  swin_attn<<<dim3(NWIN, HEADS), 32, 0, stream>>>(qb, kb, vT, bm, oh);
  swin_proj_gemm<<<NWIN, 256, 0, stream>>>(oh, pwh, proj_b, out);
}
